// MultiHeadAttentionLayer_41059887350104
// MI455X (gfx1250) — matrix operand order repaired and checked by reading, not compiled, not run on hardware
//
#include <hip/hip_runtime.h>
#include <hip/hip_bf16.h>

#define HEADS   8
#define IN_DIM  256
#define OUT_DIM 64
#define NSEQ    4096
#define QTILES  (NSEQ / 16)
#define QBLK    4                      // query tiles per attention workgroup

typedef __attribute__((ext_vector_type(16))) _Float16 v16h;
typedef __attribute__((ext_vector_type(8)))  float    v8f;
typedef __attribute__((ext_vector_type(4)))  int      v4i;

// ---------------------------------------------------------------------------
// Async global->LDS staging (gfx1250 GLOBAL_LOAD_ASYNC_TO_LDS_B128, ASYNCcnt)
// ---------------------------------------------------------------------------
#if defined(__has_builtin)
#if __has_builtin(__builtin_amdgcn_global_load_async_to_lds_b128)
#define HAVE_ASYNC_LDS 1
#endif
#endif

#ifdef HAVE_ASYNC_LDS
typedef __attribute__((address_space(1))) v4i* gas_ptr;   // global 16B vector
typedef __attribute__((address_space(3))) v4i* las_ptr;   // LDS    16B vector
__device__ __forceinline__ void async_cp16(const void* g, void* l) {
    __builtin_amdgcn_global_load_async_to_lds_b128(
        (gas_ptr)(unsigned long long)g,
        (las_ptr)(unsigned int)(unsigned long long)l, 0, 0);
}
#if __has_builtin(__builtin_amdgcn_s_wait_asynccnt)
#define WAIT_ASYNC(n) __builtin_amdgcn_s_wait_asynccnt(n)
#else
#define WAIT_ASYNC(n) asm volatile("s_wait_asynccnt %0" ::"n"(n) : "memory")
#endif
#else
// fallback: synchronous 16-byte copy through VGPRs
__device__ __forceinline__ void async_cp16(const void* g, void* l) {
    *(uint4*)l = *(const uint4*)g;
}
#define WAIT_ASYNC(n) do {} while (0)
#endif

// ---------------------------------------------------------------------------
// WMMA wrapper: D = A(16x32 f16) * B(32x16 f16) + C(16x16 f32)
// ---------------------------------------------------------------------------
__device__ __forceinline__ v8f wmma_f16(v16h a, v16h b, v8f c) {
    return __builtin_amdgcn_wmma_f32_16x16x32_f16(
        false, a, false, b, (short)0, c, false, false);
}

// A-fragment (16x32, f16) from row-major f16 source, leading dim ld.
__device__ __forceinline__ v16h load_a_f16(const _Float16* base, int ld, int lane) {
    const int m  = lane & 15;
    const int kb = (lane >> 4) * 8;
    const _Float16* p = base + m * ld + kb;
    v16h a;
#pragma unroll
    for (int j = 0; j < 8; ++j) { a[j] = p[j]; a[8 + j] = p[16 + j]; }
    return a;
}

// A-fragment (16x32) with on-the-fly f32 -> f16 conversion (for input X).
__device__ __forceinline__ v16h load_a_f32(const float* base, int ld, int lane) {
    const int m  = lane & 15;
    const int kb = (lane >> 4) * 8;
    const float* p = base + m * ld + kb;
    v16h a;
#pragma unroll
    for (int j = 0; j < 8; ++j) {
        a[j]     = (_Float16)p[j];
        a[8 + j] = (_Float16)p[16 + j];
    }
    return a;
}

// B-fragment (32x16, f16): logical B[k][n] stored as S[n][k] row-major.
// lane: N = lane&15, K = (lane>>4)*8 + 0..7 and 16 + (lane>>4)*8 + 0..7.
__device__ __forceinline__ v16h load_bt(const _Float16* base, int ld, int lane) {
    const int n  = lane & 15;
    const int kb = (lane >> 4) * 8;
    const _Float16* p = base + n * ld + kb;
    v16h b;
#pragma unroll
    for (int j = 0; j < 8; ++j) { b[j] = p[j]; b[8 + j] = p[16 + j]; }
    return b;
}

// ---------------------------------------------------------------------------
// Kernel 0: weight prep.  WQ/WK/WV [H,256,64] f32 -> transposed f16 [H,64,256]
//           lin_w [64,512] f32 -> f16 (layout kept: already B-contiguous)
// ---------------------------------------------------------------------------
__global__ void prep_weights(const float* __restrict__ WQ,
                             const float* __restrict__ WK,
                             const float* __restrict__ WV,
                             const float* __restrict__ lin_w,
                             _Float16* __restrict__ WQt,
                             _Float16* __restrict__ WKt,
                             _Float16* __restrict__ WVt,
                             _Float16* __restrict__ LWh) {
    const int per = HEADS * OUT_DIM * IN_DIM;  // 131072
    int idx = blockIdx.x * blockDim.x + threadIdx.x;
    if (idx < 3 * per) {
        int which = idx / per;
        int r     = idx % per;
        int h     = r / (OUT_DIM * IN_DIM);
        int rem   = r % (OUT_DIM * IN_DIM);
        int d     = rem / IN_DIM;
        int k     = rem % IN_DIM;
        const float* W = (which == 0) ? WQ : (which == 1) ? WK : WV;
        _Float16*    O = (which == 0) ? WQt : (which == 1) ? WKt : WVt;
        O[r] = (_Float16)W[((size_t)h * IN_DIM + k) * OUT_DIM + d];
    } else {
        int r = idx - 3 * per;
        if (r < OUT_DIM * HEADS * OUT_DIM)
            LWh[r] = (_Float16)lin_w[r];
    }
}

// ---------------------------------------------------------------------------
// Kernel 1: QKV projection.  One wave per (head, 16-row tile).
// Q,K -> row-major f16 [H,N,64];  V -> transposed f16 [H,64,N]
// ---------------------------------------------------------------------------
__global__ __launch_bounds__(32) void qkv_kernel(const float* __restrict__ X,
                                                 const _Float16* __restrict__ WQt,
                                                 const _Float16* __restrict__ WKt,
                                                 const _Float16* __restrict__ WVt,
                                                 _Float16* __restrict__ Qh,
                                                 _Float16* __restrict__ Kh,
                                                 _Float16* __restrict__ Vt) {
    const int lane = threadIdx.x;
    const int h    = blockIdx.x / QTILES;
    const int qt   = blockIdx.x % QTILES;

    v16h a[8];
#pragma unroll
    for (int t = 0; t < 8; ++t)
        a[t] = load_a_f32(X + (size_t)(qt * 16) * IN_DIM + 32 * t, IN_DIM, lane);

    const int n    = lane & 15;
    const int half = lane >> 4;

#pragma unroll
    for (int which = 0; which < 3; ++which) {
        const _Float16* Wt = (which == 0 ? WQt : which == 1 ? WKt : WVt)
                           + (size_t)h * OUT_DIM * IN_DIM;
        v8f c0 = {0,0,0,0,0,0,0,0}, c1 = c0, c2 = c0, c3 = c0;
#pragma unroll
        for (int t = 0; t < 8; ++t) {
            c0 = wmma_f16(a[t], load_bt(Wt + 0 * 16 * IN_DIM + 32 * t, IN_DIM, lane), c0);
            c1 = wmma_f16(a[t], load_bt(Wt + 1 * 16 * IN_DIM + 32 * t, IN_DIM, lane), c1);
            c2 = wmma_f16(a[t], load_bt(Wt + 2 * 16 * IN_DIM + 32 * t, IN_DIM, lane), c2);
            c3 = wmma_f16(a[t], load_bt(Wt + 3 * 16 * IN_DIM + 32 * t, IN_DIM, lane), c3);
        }
        if (which < 2) {
            _Float16* out = (which == 0 ? Qh : Kh) + ((size_t)h * NSEQ + qt * 16) * OUT_DIM;
#pragma unroll
            for (int r = 0; r < 8; ++r) {
                const int M = r + 8 * half;
                out[M * OUT_DIM + 0 * 16 + n] = (_Float16)c0[r];
                out[M * OUT_DIM + 1 * 16 + n] = (_Float16)c1[r];
                out[M * OUT_DIM + 2 * 16 + n] = (_Float16)c2[r];
                out[M * OUT_DIM + 3 * 16 + n] = (_Float16)c3[r];
            }
        } else {
            _Float16* out = Vt + (size_t)h * OUT_DIM * NSEQ;
#pragma unroll
            for (int r = 0; r < 8; ++r) {
                const int M = r + 8 * half;
                out[(size_t)(0 * 16 + n) * NSEQ + qt * 16 + M] = (_Float16)c0[r];
                out[(size_t)(1 * 16 + n) * NSEQ + qt * 16 + M] = (_Float16)c1[r];
                out[(size_t)(2 * 16 + n) * NSEQ + qt * 16 + M] = (_Float16)c2[r];
                out[(size_t)(3 * 16 + n) * NSEQ + qt * 16 + M] = (_Float16)c3[r];
            }
        }
    }
}

// ---------------------------------------------------------------------------
// Kernel 2: flash attention, 4 waves / block (64 queries), K/V blocks shared
// through LDS via double-buffered async global->LDS DMA (ASYNCcnt).
// ---------------------------------------------------------------------------
__global__ __launch_bounds__(32 * QBLK)
void attn_kernel(const _Float16* __restrict__ Qh,
                 const _Float16* __restrict__ Kh,
                 const _Float16* __restrict__ Vt,
                 _Float16* __restrict__ Acat) {
    __shared__ _Float16 Kbuf[2][32 * 64];      // 32 keys x 64 d   (8 KB)
    __shared__ _Float16 Vbuf[2][64 * 32];      // 64 d   x 32 keys (8 KB)
    __shared__ _Float16 Pbuf[QBLK][16 * 32];   // per-wave P staging (4 KB)

    const int tid  = threadIdx.x;
    const int lane = tid & 31;
    const int w    = tid >> 5;
    const int nqb  = QTILES / QBLK;            // 64
    const int h    = blockIdx.x / nqb;
    const int qt   = (blockIdx.x % nqb) * QBLK + w;
    const int n    = lane & 15;
    const int half = lane >> 4;

    const _Float16* Qb = Qh + ((size_t)h * NSEQ + qt * 16) * OUT_DIM;
    const _Float16* Kb = Kh + (size_t)h * NSEQ * OUT_DIM;
    const _Float16* Vb = Vt + (size_t)h * OUT_DIM * NSEQ;

    // Issue async copy of key block j into LDS buffer `buf`.
    // K block: contiguous 4 KB. V block: 64 rows of 64 B, row stride 8 KB.
    // 256 16-byte chunks each; every thread moves chunk tid and tid+128.
    auto issue = [&](int buf, int j) {
        const char* gK = (const char*)(Kb + (size_t)j * OUT_DIM);
        char*       lK = (char*)Kbuf[buf];
        char*       lV = (char*)Vbuf[buf];
#pragma unroll
        for (int s = 0; s < 2; ++s) {
            const int c = tid + 128 * s;
            async_cp16(gK + c * 16, lK + c * 16);
            const char* gV = (const char*)(Vb + (size_t)(c >> 2) * NSEQ + j) + (c & 3) * 16;
            async_cp16(gV, lV + c * 16);
        }
    };

    issue(0, 0);   // prologue: stage first key block

    const v16h aq0 = load_a_f16(Qb,      OUT_DIM, lane);
    const v16h aq1 = load_a_f16(Qb + 32, OUT_DIM, lane);

    v8f o0 = {0,0,0,0,0,0,0,0}, o1 = o0, o2 = o0, o3 = o0;
    float m[8], l[8];
    int   gq[8];
#pragma unroll
    for (int r = 0; r < 8; ++r) {
        m[r] = -1e30f; l[r] = 0.0f;
        gq[r] = qt * 16 + r + 8 * half;
    }
    const float inv_sqrt_d = 0.125f;

    for (int i = 0, j = 0; j < NSEQ; ++i, j += 32) {
        const int cur = i & 1;
        if (j + 32 < NSEQ) {               // overlap next block's DMA
            issue(cur ^ 1, j + 32);
            WAIT_ASYNC(4);                 // in-order: current block complete
        } else {
            WAIT_ASYNC(0);
        }
        __syncthreads();                   // LDS buffers visible block-wide

        const _Float16* Kl = Kbuf[cur];
        const _Float16* Vl = Vbuf[cur];

        // ---- scores: S[16 q][32 keys], 4 WMMAs from LDS fragments ----
        v8f s0 = {0,0,0,0,0,0,0,0}, s1 = s0;
        s0 = wmma_f16(aq0, load_bt(Kl,                 OUT_DIM, lane), s0);
        s0 = wmma_f16(aq1, load_bt(Kl + 32,            OUT_DIM, lane), s0);
        s1 = wmma_f16(aq0, load_bt(Kl + 16 * OUT_DIM,      OUT_DIM, lane), s1);
        s1 = wmma_f16(aq1, load_bt(Kl + 16 * OUT_DIM + 32, OUT_DIM, lane), s1);

        // ---- online softmax ----
        float p0[8], p1[8];
#pragma unroll
        for (int r = 0; r < 8; ++r) {
            float v0 = s0[r] * inv_sqrt_d;
            float v1 = s1[r] * inv_sqrt_d;
            v0 = (gq[r] == j + n)      ? -1e30f : v0;   // masked diagonal
            v1 = (gq[r] == j + 16 + n) ? -1e30f : v1;
            float mx = fmaxf(v0, v1);
#pragma unroll
            for (int off = 1; off < 16; off <<= 1)
                mx = fmaxf(mx, __shfl_xor(mx, off, 16));
            const float mn = fmaxf(m[r], mx);
            const float sc = __expf(m[r] - mn);
            m[r] = mn;
            p0[r] = __expf(v0 - mn);
            p1[r] = __expf(v1 - mn);
            float rs = p0[r] + p1[r];
#pragma unroll
            for (int off = 1; off < 16; off <<= 1)
                rs += __shfl_xor(rs, off, 16);
            l[r] = l[r] * sc + rs;
            o0[r] *= sc; o1[r] *= sc; o2[r] *= sc; o3[r] *= sc;
        }

        // ---- stage P (C layout -> row-major 16x32) in this wave's region ----
        _Float16* P = Pbuf[w];
#pragma unroll
        for (int r = 0; r < 8; ++r) {
            const int M = r + 8 * half;
            P[M * 32 + n]      = (_Float16)p0[r];
            P[M * 32 + 16 + n] = (_Float16)p1[r];
        }
        __syncthreads();                   // also orders P writes vs reads

        // ---- O += P(16x32) * V(32x16 per d-tile), 4 WMMAs ----
        const v16h pa = load_a_f16(P, 32, lane);
        o0 = wmma_f16(pa, load_bt(Vl + (0 * 16) * 32, 32, lane), o0);
        o1 = wmma_f16(pa, load_bt(Vl + (1 * 16) * 32, 32, lane), o1);
        o2 = wmma_f16(pa, load_bt(Vl + (2 * 16) * 32, 32, lane), o2);
        o3 = wmma_f16(pa, load_bt(Vl + (3 * 16) * 32, 32, lane), o3);

        __syncthreads();                   // buf[cur] is re-filled next iter
    }

    // ---- normalize and emit concatenated head output [N, H*64] f16 ----
#pragma unroll
    for (int r = 0; r < 8; ++r) {
        const float inv = 1.0f / l[r];
        const int   M   = r + 8 * half;
        _Float16* out = Acat + (size_t)(qt * 16 + M) * (HEADS * OUT_DIM) + h * OUT_DIM;
        out[0 * 16 + n] = (_Float16)(o0[r] * inv);
        out[1 * 16 + n] = (_Float16)(o1[r] * inv);
        out[2 * 16 + n] = (_Float16)(o2[r] * inv);
        out[3 * 16 + n] = (_Float16)(o3[r] * inv);
    }
}

// ---------------------------------------------------------------------------
// Kernel 3: final linear.  out[4096,64] = Acat[4096,512] @ lin_w^T + b  (f32)
// ---------------------------------------------------------------------------
__global__ __launch_bounds__(32) void out_linear(const _Float16* __restrict__ Acat,
                                                 const _Float16* __restrict__ LWh,
                                                 const float* __restrict__ bias,
                                                 float* __restrict__ out) {
    const int lane = threadIdx.x;
    const int qt   = blockIdx.x;
    const int KDIM = HEADS * OUT_DIM;   // 512

    v8f c0 = {0,0,0,0,0,0,0,0}, c1 = c0, c2 = c0, c3 = c0;
#pragma unroll
    for (int t = 0; t < KDIM / 32; ++t) {
        const v16h a = load_a_f16(Acat + (size_t)(qt * 16) * KDIM + 32 * t, KDIM, lane);
        c0 = wmma_f16(a, load_bt(LWh + (size_t)(0 * 16) * KDIM + 32 * t, KDIM, lane), c0);
        c1 = wmma_f16(a, load_bt(LWh + (size_t)(1 * 16) * KDIM + 32 * t, KDIM, lane), c1);
        c2 = wmma_f16(a, load_bt(LWh + (size_t)(2 * 16) * KDIM + 32 * t, KDIM, lane), c2);
        c3 = wmma_f16(a, load_bt(LWh + (size_t)(3 * 16) * KDIM + 32 * t, KDIM, lane), c3);
    }
    const int n = lane & 15, half = lane >> 4;
#pragma unroll
    for (int r = 0; r < 8; ++r) {
        const int M = qt * 16 + r + 8 * half;
        out[(size_t)M * OUT_DIM + 0 * 16 + n] = c0[r] + bias[0 * 16 + n];
        out[(size_t)M * OUT_DIM + 1 * 16 + n] = c1[r] + bias[1 * 16 + n];
        out[(size_t)M * OUT_DIM + 2 * 16 + n] = c2[r] + bias[2 * 16 + n];
        out[(size_t)M * OUT_DIM + 3 * 16 + n] = c3[r] + bias[3 * 16 + n];
    }
}

// ---------------------------------------------------------------------------
extern "C" void kernel_launch(void* const* d_in, const int* in_sizes, int n_in,
                              void* d_out, int out_size, void* d_ws, size_t ws_size,
                              hipStream_t stream) {
    const float* X     = (const float*)d_in[0];   // [4096,256]
    const float* WQ    = (const float*)d_in[1];   // [8,256,64]
    const float* WK    = (const float*)d_in[2];
    const float* WV    = (const float*)d_in[3];
    const float* lin_w = (const float*)d_in[4];   // [64,512]
    const float* lin_b = (const float*)d_in[5];   // [64]
    float*       out   = (float*)d_out;           // [4096,64]

    char* ws = (char*)d_ws;
    _Float16* WQt  = (_Float16*)ws; ws += (size_t)HEADS * OUT_DIM * IN_DIM * 2;
    _Float16* WKt  = (_Float16*)ws; ws += (size_t)HEADS * OUT_DIM * IN_DIM * 2;
    _Float16* WVt  = (_Float16*)ws; ws += (size_t)HEADS * OUT_DIM * IN_DIM * 2;
    _Float16* LWh  = (_Float16*)ws; ws += (size_t)OUT_DIM * HEADS * OUT_DIM * 2;
    _Float16* Qh   = (_Float16*)ws; ws += (size_t)HEADS * NSEQ * OUT_DIM * 2;
    _Float16* Kh   = (_Float16*)ws; ws += (size_t)HEADS * NSEQ * OUT_DIM * 2;
    _Float16* Vt   = (_Float16*)ws; ws += (size_t)HEADS * OUT_DIM * NSEQ * 2;
    _Float16* Acat = (_Float16*)ws; ws += (size_t)NSEQ * HEADS * OUT_DIM * 2;

    const int prep_elems = 3 * HEADS * OUT_DIM * IN_DIM + OUT_DIM * HEADS * OUT_DIM;
    prep_weights<<<(prep_elems + 255) / 256, 256, 0, stream>>>(WQ, WK, WV, lin_w,
                                                               WQt, WKt, WVt, LWh);

    qkv_kernel<<<HEADS * QTILES, 32, 0, stream>>>(X, WQt, WKt, WVt, Qh, Kh, Vt);

    attn_kernel<<<HEADS * (QTILES / QBLK), 32 * QBLK, 0, stream>>>(Qh, Kh, Vt, Acat);

    out_linear<<<QTILES, 32, 0, stream>>>(Acat, LWh, lin_b, out);
}
